// DFLM_67319317397798
// MI455X (gfx1250) — matrix operand order repaired and checked by reading, not compiled, not run on hardware
//
#include <hip/hip_runtime.h>
#include <hip/hip_bf16.h>
#include <cstdint>

typedef __attribute__((ext_vector_type(16))) _Float16 v16h;
typedef __attribute__((ext_vector_type(8)))  float    v8f;

#define WMMA_F32_F16(a, b, c) \
  __builtin_amdgcn_wmma_f32_16x16x32_f16(false, (a), false, (b), (short)0, (c), false, false)

__device__ __forceinline__ v8f v8f_zero() {
  v8f z = {0.f, 0.f, 0.f, 0.f, 0.f, 0.f, 0.f, 0.f};
  return z;
}

// CDNA5 async copy: global -> LDS, 16 bytes per lane, tracked by ASYNCcnt.
__device__ __forceinline__ void async_load_b128(unsigned lds_off, unsigned gbyte_off,
                                                const void* sbase) {
  asm volatile("global_load_async_to_lds_b128 %0, %1, %2"
               :: "v"(lds_off), "v"(gbyte_off), "s"(sbase) : "memory");
}
__device__ __forceinline__ void wait_async0() {
  asm volatile("s_wait_asynccnt 0" ::: "memory");
}

// A-matrix fragment (16x32 f16, MxK). ISA 7.12.2: lane half h=lane>>4, row m=lane&15.
// VGPR j holds K = (j>>2)*16 + h*8 + (j&3)*2 (+0,+1). base is [row][k] with `stride` f16/row.
__device__ __forceinline__ v16h load_a_frag(const _Float16* base, int stride, int row0, int lane) {
  const int h = lane >> 4, mr = lane & 15;
  union { v16h v; unsigned u[8]; } f;
  const _Float16* rp = base + (size_t)(row0 + mr) * stride;
#pragma unroll
  for (int j = 0; j < 8; ++j) {
    const int kb = (j >> 2) * 16 + h * 8 + (j & 3) * 2;
    f.u[j] = *(const unsigned*)(rp + kb);
  }
  return f.v;
}

// B-matrix fragment (32x16 f16, KxN). Lane half h, col n=lane&15.
// VGPR j holds K = (j>>2)*16 + h*8 + (j&3)*2 (+0,+1). base is TRANSPOSED LDS [n][k], `stride` f16 per n.
__device__ __forceinline__ v16h load_b_frag(const _Float16* base, int stride, int col0, int lane) {
  const int h = lane >> 4, nr = lane & 15;
  union { v16h v; unsigned u[8]; } f;
  const _Float16* cp = base + (size_t)(col0 + nr) * stride;
#pragma unroll
  for (int j = 0; j < 8; ++j)
    f.u[j] = *(const unsigned*)(cp + (j >> 2) * 16 + h * 8 + (j & 3) * 2);
  return f.v;
}

// ---------------------------------------------------------------------------
// Generic GEMM: Out[M,Nout] = A[M,K] @ W[K,Nout] + bias.
// Block tile 128(M) x 64(N), 4 waves in 2x2; wave tile 64x32 = 4x2 WMMA.
// Double-buffered LDS staging (f32 -> f16 conversion on the fly).
// K % 32 == 0, M % 128 == 0, Nout % 64 == 0.
// ---------------------------------------------------------------------------
template <typename AT>
__device__ __forceinline__ void stage_tile(const AT* __restrict__ A,
                                           const float* __restrict__ W,
                                           _Float16 (*sA)[34], _Float16 (*sB)[34],
                                           int mBase, int nBase, int K, int Nout,
                                           int k0, int tid) {
#pragma unroll 4
  for (int i = 0; i < 32; ++i) {  // A: 128 x 32
    int e = tid + i * 128;
    int r = e >> 5, c = e & 31;
    sA[r][c] = (_Float16)(float)A[(size_t)(mBase + r) * K + (k0 + c)];
  }
#pragma unroll 4
  for (int i = 0; i < 16; ++i) {  // W: 32 x 64, stored transposed [n][k]
    int e = tid + i * 128;
    int kk = e >> 6, nn = e & 63;
    sB[nn][kk] = (_Float16)W[(size_t)(k0 + kk) * Nout + (nBase + nn)];
  }
}

template <typename AT, bool OUT_F32>
__global__ __launch_bounds__(128) void gemm_ws_kernel(
    const AT* __restrict__ A, const float* __restrict__ W,
    const float* __restrict__ bias, void* __restrict__ outv,
    int M, int K, int Nout) {
  __shared__ _Float16 sA[2][128][34];
  __shared__ _Float16 sB[2][64][34];
  const int tid = threadIdx.x, lane = tid & 31, w = tid >> 5;
  const int wm = w >> 1, wn = w & 1;
  const int mBase = blockIdx.y * 128, nBase = blockIdx.x * 64;
  (void)M;

  v8f acc[4][2];
#pragma unroll
  for (int mi = 0; mi < 4; ++mi)
#pragma unroll
    for (int ni = 0; ni < 2; ++ni) acc[mi][ni] = v8f_zero();

  const int KT = K >> 5;
  stage_tile(A, W, sA[0], sB[0], mBase, nBase, K, Nout, 0, tid);

  for (int kt = 0; kt < KT; ++kt) {
    __syncthreads();  // staging of tile kt complete; buffer kt&1 readable
    if (kt + 1 < KT)
      stage_tile(A, W, sA[(kt + 1) & 1], sB[(kt + 1) & 1],
                 mBase, nBase, K, Nout, (kt + 1) * 32, tid);
    const int cur = kt & 1;
    v16h af[4], bf[2];
#pragma unroll
    for (int mi = 0; mi < 4; ++mi)
      af[mi] = load_a_frag(&sA[cur][0][0], 34, wm * 64 + mi * 16, lane);
#pragma unroll
    for (int ni = 0; ni < 2; ++ni)
      bf[ni] = load_b_frag(&sB[cur][0][0], 34, wn * 32 + ni * 16, lane);
#pragma unroll
    for (int mi = 0; mi < 4; ++mi)
#pragma unroll
      for (int ni = 0; ni < 2; ++ni)
        acc[mi][ni] = WMMA_F32_F16(af[mi], bf[ni], acc[mi][ni]);
  }

  // C/D layout: VGPR r -> row r (lanes 0-15) or 8+r (lanes 16-31); col = lane&15.
  const int h = lane >> 4, nr = lane & 15;
#pragma unroll
  for (int mi = 0; mi < 4; ++mi)
#pragma unroll
    for (int ni = 0; ni < 2; ++ni) {
      const int col = nBase + wn * 32 + ni * 16 + nr;
      const float bv = bias[col];
#pragma unroll
      for (int r = 0; r < 8; ++r) {
        const int row = mBase + wm * 64 + mi * 16 + h * 8 + r;
        const float val = acc[mi][ni][r] + bv;
        if (OUT_F32)
          ((float*)outv)[(size_t)row * Nout + col] = val;
        else
          ((_Float16*)outv)[(size_t)row * Nout + col] = (_Float16)val;
      }
    }
}

// ---------------------------------------------------------------------------
// Branch B: flash attention, agent queries (64) vs pixel keys (4096), per (b,h).
// logits = (qa1.kh1)*wb0*S + (qa2.kh2)*wb1*S  (folded into 64-dim concat query).
// Key tile staged via global_load_async_to_lds_b128 (ASYNCcnt path).
// ---------------------------------------------------------------------------
__global__ __launch_bounds__(128) void branchB_kernel(
    const _Float16* __restrict__ qag, const _Float16* __restrict__ khf,
    const _Float16* __restrict__ vhf, const float* __restrict__ wb,
    _Float16* __restrict__ xs) {
  const int b = blockIdx.y, hd = blockIdx.x;
  const int tid = threadIdx.x, lane = tid & 31, w = tid >> 5;
  __shared__ _Float16 sQ[64][66];                    // [agent][cat-dim] pre-scaled
  __shared__ __align__(16) _Float16 sK[128][72];     // [key][cat-dim]; 144B rows (16B-mult)
  __shared__ _Float16 sV[32][130];                   // [dim][key]
  __shared__ _Float16 sP[4][16][130];                // per-wave P [row][key]

  const float SCALE = 0.17677669529663687f;  // 32^-0.5
  const float sc1 = wb[0] * SCALE, sc2 = wb[1] * SCALE;

  for (int i = 0; i < 32; ++i) {
    int e = tid + i * 128;
    int row = e >> 6, dd = e & 63;
    size_t base = ((size_t)b * 64 + row) * 768;
    float v = (dd < 32) ? (float)qag[base + hd * 32 + dd] * sc1
                        : (float)qag[base + 384 + hd * 32 + (dd - 32)] * sc2;
    sQ[row][dd] = (_Float16)v;
  }
  __syncthreads();

  const v16h aq0 = load_a_frag(&sQ[0][0], 66, w * 16, lane);
  const v16h aq1 = load_a_frag(&sQ[0][32], 66, w * 16, lane);

  const _Float16* kbase = khf + (size_t)b * 4096 * 768;  // uniform -> SGPR pair
  const unsigned sKoff = (unsigned)(uintptr_t)&sK[0][0];

  float mrow[8], lrow[8];
  v8f accO[2] = {v8f_zero(), v8f_zero()};
#pragma unroll
  for (int r = 0; r < 8; ++r) { mrow[r] = -1e30f; lrow[r] = 0.f; }

  for (int t0 = 0; t0 < 4096; t0 += 128) {
    // Async-stage keys: 128 keys x 2 segments x 64B = 16KB, B128 per lane.
#pragma unroll
    for (int i = 0; i < 8; ++i) {
      int c = tid + i * 128;             // chunk id 0..1023
      int key = c >> 3, sub = c & 7;
      int seg = sub >> 2, q = sub & 3;   // seg: branch-1/2 halves, q: 16B chunk
      unsigned goff = (unsigned)((((t0 + key) * 768) + seg * 384 + hd * 32 + q * 8) * 2);
      unsigned loff = sKoff + (unsigned)((key * 72 + seg * 32 + q * 8) * 2);
      async_load_b128(loff, goff, kbase);
    }
    // V tile needs a transpose -> VGPR staging.
    for (int i = 0; i < 32; ++i) {
      int e = tid + i * 128;
      int key = e >> 5, dd = e & 31;
      sV[dd][key] = vhf[((size_t)b * 4096 + t0 + key) * 384 + hd * 32 + dd];
    }
    wait_async0();
    __syncthreads();

    v8f s[8];
#pragma unroll
    for (int nt = 0; nt < 8; ++nt) {
      s[nt] = WMMA_F32_F16(aq0, load_b_frag(&sK[0][0], 72, nt * 16, lane), v8f_zero());
      s[nt] = WMMA_F32_F16(aq1, load_b_frag(&sK[0][32], 72, nt * 16, lane), s[nt]);
    }

    // online softmax (rows split: lanes 0-15 -> row r, lanes 16-31 -> row 8+r)
#pragma unroll
    for (int r = 0; r < 8; ++r) {
      float tm = s[0][r];
#pragma unroll
      for (int nt = 1; nt < 8; ++nt) tm = fmaxf(tm, s[nt][r]);
#pragma unroll
      for (int off = 8; off; off >>= 1) tm = fmaxf(tm, __shfl_xor(tm, off, 32));
      const float mn = fmaxf(mrow[r], tm);
      const float scl = __expf(mrow[r] - mn);
      mrow[r] = mn;
      lrow[r] *= scl;
      accO[0][r] *= scl;
      accO[1][r] *= scl;
      float acc = 0.f;
      const int rloc = (lane >> 4) * 8 + r;
#pragma unroll
      for (int nt = 0; nt < 8; ++nt) {
        const float p = __expf(s[nt][r] - mn);
        acc += p;
        sP[w][rloc][nt * 16 + (lane & 15)] = (_Float16)p;
      }
#pragma unroll
      for (int off = 8; off; off >>= 1) acc += __shfl_xor(acc, off, 32);
      lrow[r] += acc;
    }

    // O += P @ V  (K = 128 keys, chained in 4 chunks of 32)
#pragma unroll
    for (int nt2 = 0; nt2 < 2; ++nt2)
#pragma unroll
      for (int kc = 0; kc < 4; ++kc) {
        v16h ap = load_a_frag(&sP[w][0][kc * 32], 130, 0, lane);
        v16h bv = load_b_frag(&sV[0][kc * 32], 130, nt2 * 16, lane);
        accO[nt2] = WMMA_F32_F16(ap, bv, accO[nt2]);
      }
    __syncthreads();
  }

  const int h = lane >> 4, nr = lane & 15;
#pragma unroll
  for (int nt2 = 0; nt2 < 2; ++nt2)
#pragma unroll
    for (int r = 0; r < 8; ++r) {
      const int agent = w * 16 + h * 8 + r;
      const float val = accO[nt2][r] / lrow[r];
      xs[(((size_t)b * 12 + hd) * 64 + agent) * 32 + nt2 * 16 + nr] = (_Float16)val;
    }
}

// ---------------------------------------------------------------------------
// Branch A: pixel queries vs 64 agent keys, softmax over agents, @ x_s.
// Block = 8 waves, 128 pixel rows per block; per (b,h,pixel-tile).
// ---------------------------------------------------------------------------
__global__ __launch_bounds__(256) void branchA_kernel(
    const _Float16* __restrict__ qlf, const _Float16* __restrict__ kag,
    const _Float16* __restrict__ xs, const float* __restrict__ wa,
    _Float16* __restrict__ xout) {
  const int b = blockIdx.z, hd = blockIdx.y, pt = blockIdx.x;
  const int tid = threadIdx.x, lane = tid & 31, w = tid >> 5;
  __shared__ _Float16 sKA[64][66];      // [agent][cat-dim]
  __shared__ _Float16 sXS[32][66];      // [dim][agent]
  __shared__ _Float16 sP[8][16][66];    // per-wave P [row][agent]
  const float SCALE = 0.17677669529663687f;
  const float sc1 = wa[0] * SCALE, sc2 = wa[1] * SCALE;

  for (int i = 0; i < 16; ++i) {
    int e = tid + i * 256;
    int ag = e >> 6, dd = e & 63;
    size_t base = ((size_t)b * 64 + ag) * 768;
    sKA[ag][dd] = (dd < 32) ? kag[base + hd * 32 + dd]
                            : kag[base + 384 + hd * 32 + (dd - 32)];
  }
  for (int i = 0; i < 8; ++i) {
    int e = tid + i * 256;
    int ag = e & 63, dd = e >> 6;
    sXS[dd][ag] = xs[(((size_t)b * 12 + hd) * 64 + ag) * 32 + dd];
  }
  __syncthreads();

  const int row0 = pt * 128 + w * 16;
  v16h aq0, aq1;  // scaled concat queries straight from global, A layout
  {
    const int h = lane >> 4, mr = lane & 15;
    union { v16h v; _Float16 x[16]; } f0, f1;
    const size_t rbase = ((size_t)b * 4096 + row0 + mr) * 768;
#pragma unroll
    for (int j = 0; j < 8; ++j) {
      const int kb = (j >> 2) * 16 + h * 8 + (j & 3) * 2;
      f0.x[2 * j]     = (_Float16)((float)qlf[rbase + hd * 32 + kb] * sc1);
      f0.x[2 * j + 1] = (_Float16)((float)qlf[rbase + hd * 32 + kb + 1] * sc1);
      f1.x[2 * j]     = (_Float16)((float)qlf[rbase + 384 + hd * 32 + kb] * sc2);
      f1.x[2 * j + 1] = (_Float16)((float)qlf[rbase + 384 + hd * 32 + kb + 1] * sc2);
    }
    aq0 = f0.v;
    aq1 = f1.v;
  }

  v8f s[4];
#pragma unroll
  for (int nt = 0; nt < 4; ++nt) {
    s[nt] = WMMA_F32_F16(aq0, load_b_frag(&sKA[0][0], 66, nt * 16, lane), v8f_zero());
    s[nt] = WMMA_F32_F16(aq1, load_b_frag(&sKA[0][32], 66, nt * 16, lane), s[nt]);
  }

  float linv[8];
#pragma unroll
  for (int r = 0; r < 8; ++r) {
    float tm = s[0][r];
#pragma unroll
    for (int nt = 1; nt < 4; ++nt) tm = fmaxf(tm, s[nt][r]);
#pragma unroll
    for (int off = 8; off; off >>= 1) tm = fmaxf(tm, __shfl_xor(tm, off, 32));
    float acc = 0.f;
    const int rloc = (lane >> 4) * 8 + r;
#pragma unroll
    for (int nt = 0; nt < 4; ++nt) {
      const float p = __expf(s[nt][r] - tm);
      acc += p;
      sP[w][rloc][nt * 16 + (lane & 15)] = (_Float16)p;
    }
#pragma unroll
    for (int off = 8; off; off >>= 1) acc += __shfl_xor(acc, off, 32);
    linv[r] = 1.f / acc;
  }

  v8f accO[2] = {v8f_zero(), v8f_zero()};
#pragma unroll
  for (int nt2 = 0; nt2 < 2; ++nt2)
#pragma unroll
    for (int kc = 0; kc < 2; ++kc) {
      v16h ap = load_a_frag(&sP[w][0][kc * 32], 66, 0, lane);
      v16h bx = load_b_frag(&sXS[0][kc * 32], 66, nt2 * 16, lane);
      accO[nt2] = WMMA_F32_F16(ap, bx, accO[nt2]);
    }

  const int h = lane >> 4, nr = lane & 15;
#pragma unroll
  for (int nt2 = 0; nt2 < 2; ++nt2)
#pragma unroll
    for (int r = 0; r < 8; ++r) {
      const int row = row0 + h * 8 + r;
      xout[((size_t)b * 4096 + row) * 384 + hd * 32 + nt2 * 16 + nr] =
          (_Float16)(accO[nt2][r] * linv[r]);
    }
}

// ---------------------------------------------------------------------------
extern "C" void kernel_launch(void* const* d_in, const int* in_sizes, int n_in,
                              void* d_out, int out_size, void* d_ws, size_t ws_size,
                              hipStream_t stream) {
  (void)in_sizes; (void)n_in; (void)out_size; (void)ws_size;
  const float* x      = (const float*)d_in[0];
  const float* attn   = (const float*)d_in[1];
  const float* agent  = (const float*)d_in[2];
  const float* Wq_lf  = (const float*)d_in[3];
  const float* bq_lf  = (const float*)d_in[4];
  const float* Wk_ag  = (const float*)d_in[5];
  const float* bk_ag  = (const float*)d_in[6];
  const float* wa     = (const float*)d_in[7];
  // d_in[8] = ba : uniform logit shift, softmax-invariant -> unused
  const float* Wq_ag  = (const float*)d_in[9];
  const float* bq_ag  = (const float*)d_in[10];
  const float* Wk_hf  = (const float*)d_in[11];
  const float* bk_hf  = (const float*)d_in[12];
  const float* Wv_hf  = (const float*)d_in[13];
  const float* bv_hf  = (const float*)d_in[14];
  const float* wb     = (const float*)d_in[15];
  // d_in[16] = bb : softmax-invariant -> unused
  const float* Wproj  = (const float*)d_in[17];
  const float* bproj  = (const float*)d_in[18];

  char* ws = (char*)d_ws;
  size_t off = 0;
  auto carve = [&](size_t nelem) -> _Float16* {
    _Float16* p = (_Float16*)(ws + off);
    off += ((nelem * sizeof(_Float16) + 255) / 256) * 256;
    return p;
  };
  _Float16* qlf  = carve(32768ull * 768);  // x @ Wq_lf        [B*N, 2C]
  _Float16* kag  = carve(512ull * 768);    // agent @ Wk_ag    [B*Na, 2C]
  _Float16* qag  = carve(512ull * 768);    // agent @ Wq_ag    [B*Na, 2C]
  _Float16* khf  = carve(32768ull * 768);  // attn @ Wk_hf     [B*N, 2C]
  _Float16* vhf  = carve(32768ull * 384);  // attn @ Wv_hf     [B*N, C]
  _Float16* xsb  = carve(8ull * 12 * 64 * 32);  // x_s [B,H,Na,D]
  _Float16* xout = carve(32768ull * 384);  // attn_agent @ x_s [B*N, C]

  const dim3 blk128(128);
  gemm_ws_kernel<float, false><<<dim3(12, 256), blk128, 0, stream>>>(x,     Wq_lf, bq_lf, (void*)qlf, 32768, 384, 768);
  gemm_ws_kernel<float, false><<<dim3(12, 4),   blk128, 0, stream>>>(agent, Wk_ag, bk_ag, (void*)kag, 512,   384, 768);
  gemm_ws_kernel<float, false><<<dim3(12, 4),   blk128, 0, stream>>>(agent, Wq_ag, bq_ag, (void*)qag, 512,   384, 768);
  gemm_ws_kernel<float, false><<<dim3(12, 256), blk128, 0, stream>>>(attn,  Wk_hf, bk_hf, (void*)khf, 32768, 384, 768);
  gemm_ws_kernel<float, false><<<dim3(6, 256),  blk128, 0, stream>>>(attn,  Wv_hf, bv_hf, (void*)vhf, 32768, 384, 384);

  branchB_kernel<<<dim3(12, 8), blk128, 0, stream>>>(qag, khf, vhf, wb, xsb);
  branchA_kernel<<<dim3(32, 12, 8), dim3(256), 0, stream>>>(qlf, kag, xsb, wa, xout);

  gemm_ws_kernel<_Float16, true><<<dim3(6, 256), blk128, 0, stream>>>(xout, Wproj, bproj, d_out, 32768, 384, 384);
}
